// SNNModel_79929341378954
// MI455X (gfx1250) — matrix operand order repaired and checked by reading, not compiled, not run on hardware
//
#include <hip/hip_runtime.h>

// ---------------- problem constants (fixed by setup_inputs) ----------------
#define NB   4096   // batch
#define ND   512    // input features
#define NH   1024   // hidden
#define BETA_F 0.9f
#define TH_F   1.0f

typedef __attribute__((ext_vector_type(16))) _Float16 v16h;
typedef __attribute__((ext_vector_type(8)))  _Float16 v8h;
typedef __attribute__((ext_vector_type(8)))  float    v8f;
typedef int v4i __attribute__((vector_size(16)));   // 16B chunk for async copy

// ---- gfx1250 async global->LDS copy (ASYNCcnt-tracked), with fallback ----
#if __has_builtin(__builtin_amdgcn_global_load_async_to_lds_b128)
#define USE_ASYNC_LDS 1
#else
#define USE_ASYNC_LDS 0
#endif

#if USE_ASYNC_LDS
__device__ __forceinline__ void async_cp16B(const _Float16* g, _Float16* l) {
    // copies 16 bytes per lane, tracked by ASYNCcnt
    __builtin_amdgcn_global_load_async_to_lds_b128(
        (__attribute__((address_space(1))) v4i*)(g),
        (__attribute__((address_space(3))) v4i*)(l), 0, 0);
}
#if __has_builtin(__builtin_amdgcn_s_wait_asynccnt)
#define WAIT_ASYNC(n) __builtin_amdgcn_s_wait_asynccnt(n)
#else
#define WAIT_ASYNC(n) asm volatile("s_wait_asynccnt %0" ::"n"(n))
#endif
#endif

// ---------------- BatchNorm batch statistics -> scale/shift ----------------
__global__ __launch_bounds__(256) void snn_bn_stats(
    const float* __restrict__ x, const float* __restrict__ gamma,
    const float* __restrict__ bnb, float* __restrict__ scale,
    float* __restrict__ shift)
{
    __shared__ float s_sum[256];
    __shared__ float s_sq[256];
    const int d = blockIdx.x;
    float sum = 0.f, sq = 0.f;
    for (int b = threadIdx.x; b < NB; b += 256) {
        float v = x[(size_t)b * ND + d];
        sum += v; sq += v * v;
    }
    s_sum[threadIdx.x] = sum; s_sq[threadIdx.x] = sq;
    __syncthreads();
    for (int off = 128; off > 0; off >>= 1) {
        if (threadIdx.x < off) {
            s_sum[threadIdx.x] += s_sum[threadIdx.x + off];
            s_sq[threadIdx.x]  += s_sq[threadIdx.x + off];
        }
        __syncthreads();
    }
    if (threadIdx.x == 0) {
        float mu  = s_sum[0] * (1.0f / NB);
        float var = s_sq[0] * (1.0f / NB) - mu * mu;   // jnp.var: biased
        float sc  = gamma[d] * rsqrtf(var + 1e-5f);
        scale[d]  = sc;
        shift[d]  = bnb[d] - mu * sc;
    }
}

// x_norm -> f16 (A operand of GEMM1)
__global__ __launch_bounds__(256) void snn_norm_to_f16(
    const float* __restrict__ x, const float* __restrict__ scale,
    const float* __restrict__ shift, _Float16* __restrict__ a16)
{
    int i = blockIdx.x * 256 + threadIdx.x;
    if (i < NB * ND) {
        int d = i & (ND - 1);
        a16[i] = (_Float16)(x[i] * scale[d] + shift[d]);
    }
}

// generic f32 -> f16 conversion (weights)
__global__ __launch_bounds__(256) void snn_cvt_f16(
    const float* __restrict__ src, _Float16* __restrict__ dst, int n)
{
    int i = blockIdx.x * 256 + threadIdx.x;
    if (i < n) dst[i] = (_Float16)src[i];
}

// ---------------- WMMA GEMM: C(MxN) = A(MxK) * Bt(NxK)^T + bias -----------
// Block = 8 waves, 128x128 output tile; wave = 16 rows x 128 cols (8 acc tiles).
// Async path: double-buffered 128x64 A/B tiles staged to LDS with
// GLOBAL_LOAD_ASYNC_TO_LDS_B128; fragments read from LDS (ds_load_b128) and
// software-pipelined one WMMA ahead so ds latency is not exposed per-WMMA.
// FUSE_LIF: epilogue does the leaky-integrate-fire update on `mem` and
// emits {0,1} spikes (f16) instead of storing C.
template <bool FUSE_LIF>
__global__ __launch_bounds__(256) void snn_wmma_gemm(
    const _Float16* __restrict__ A,   // M x K
    const _Float16* __restrict__ Bt,  // N x K
    const float*    __restrict__ bias,// N
    float*          __restrict__ C,   // M x N   (used when !FUSE_LIF)
    float*          __restrict__ mem, // M x N   (used when FUSE_LIF)
    _Float16*       __restrict__ spk, // M x N   (used when FUSE_LIF)
    int M, int N, int K)
{
    const int lane  = threadIdx.x & 31;
    const int wave  = threadIdx.x >> 5;
    const int mBase = blockIdx.y * 128 + wave * 16;
    const int nBase = blockIdx.x * 128;
    const int l16   = lane & 15;   // row (A) / col (B) within a 16x16 tile
    const int kh    = lane >> 4;   // K-half select

    v8f acc[8];
#pragma unroll
    for (int t = 0; t < 8; ++t) acc[t] = (v8f){};

#if USE_ASYNC_LDS
    // ---- async double-buffered LDS pipeline ----
    constexpr int KC = 64;         // K-chunk per stage
    constexpr int SA = 72;         // padded row stride (halves): 144B, 16B-aligned
    __shared__ _Float16 smA[2][128 * SA];
    __shared__ _Float16 smB[2][128 * SA];

    const int mBlock = blockIdx.y * 128;
    const int nBlock = blockIdx.x * 128;
    const int S = K / KC;

    auto stage_load = [&](int s) {
        const int buf = s & 1;
        const int kc0 = s * KC;
#pragma unroll
        for (int c = 0; c < 4; ++c) {           // A tile: 128 x 64 halves
            int q   = threadIdx.x + 256 * c;    // 1024 chunks of 8 halves
            int row = q >> 3, col = (q & 7) * 8;
            async_cp16B(A + (size_t)(mBlock + row) * K + kc0 + col,
                        &smA[buf][row * SA + col]);
        }
#pragma unroll
        for (int c = 0; c < 4; ++c) {           // B tile: 128 x 64 halves
            int q   = threadIdx.x + 256 * c;
            int row = q >> 3, col = (q & 7) * 8;
            async_cp16B(Bt + (size_t)(nBlock + row) * K + kc0 + col,
                        &smB[buf][row * SA + col]);
        }
    };

    // A fragment: lane (kh*16+m) holds K=16*kh+0..15
    auto loadA = [&](const _Float16* As, int kb) -> v16h {
        const _Float16* ar = As + (wave * 16 + l16) * SA + kb;
        v8h lo = *(const v8h*)(ar + 16 * kh);
        v8h hi = *(const v8h*)(ar + 16 * kh + 8);
        v16h r;
#pragma unroll
        for (int i = 0; i < 8; ++i) { r[i] = lo[i]; r[i + 8] = hi[i]; }
        return r;
    };
    // B fragment: lane (kh*16+n) holds K=16*kh+0..15 of column n
    auto loadB = [&](const _Float16* Bs, int kb, int t) -> v16h {
        const _Float16* br = Bs + (16 * t + l16) * SA + kb + 16 * kh;
        v8h lo = *(const v8h*)(br);
        v8h hi = *(const v8h*)(br + 8);
        v16h r;
#pragma unroll
        for (int i = 0; i < 8; ++i) { r[i] = lo[i]; r[i + 8] = hi[i]; }
        return r;
    };

    stage_load(0);
    if (S > 1) stage_load(1);

    for (int s = 0; s < S; ++s) {
        if (s + 1 < S) { WAIT_ASYNC(8); }       // stage s landed, s+1 in flight
        else           { WAIT_ASYNC(0); }
        __syncthreads();

        const _Float16* As = smA[s & 1];
        const _Float16* Bs = smB[s & 1];

        // software-pipelined: next fragment loads issue before current WMMA
        v16h aNext = loadA(As, 0);
        v16h bNext = loadB(Bs, 0, 0);
#pragma unroll
        for (int kb = 0; kb < KC; kb += 32) {
            v16h a = aNext;
#pragma unroll
            for (int t = 0; t < 8; ++t) {
                v16h b = bNext;
                if (t < 7) {
                    bNext = loadB(Bs, kb, t + 1);
                } else if (kb + 32 < KC) {
                    aNext = loadA(As, kb + 32);
                    bNext = loadB(Bs, kb + 32, 0);
                }
                acc[t] = __builtin_amdgcn_wmma_f32_16x16x32_f16(
                    false, a, false, b, (short)0, acc[t], false, false);
            }
        }
        __syncthreads();                        // buf free before overwrite
        if (s + 2 < S) stage_load(s + 2);
    }
#else
    // ---- fallback: direct-from-global fragments ----
    const _Float16* Arow = A + (size_t)(mBase + l16) * K;
    for (int kb = 0; kb < K; kb += 32) {
        v8h alo = *(const v8h*)(Arow + kb + 16 * kh);
        v8h ahi = *(const v8h*)(Arow + kb + 16 * kh + 8);
        v16h a;
#pragma unroll
        for (int i = 0; i < 8; ++i) { a[i] = alo[i]; a[i + 8] = ahi[i]; }
        __builtin_prefetch(Arow + kb + 32, 0, 0);
#pragma unroll
        for (int t = 0; t < 8; ++t) {
            const _Float16* Brow =
                Bt + (size_t)(nBase + 16 * t + l16) * K + kb + 16 * kh;
            v16h b = *(const v16h*)Brow;
            acc[t] = __builtin_amdgcn_wmma_f32_16x16x32_f16(
                false, a, false, b, (short)0, acc[t], false, false);
        }
    }
#endif

    // C/D layout: lane l -> n = l&15, m = vgpr + 8*(l>>4)
#pragma unroll
    for (int t = 0; t < 8; ++t) {
        const int n = nBase + 16 * t + l16;
        const float bv = bias[n];
#pragma unroll
        for (int r = 0; r < 8; ++r) {
            const int m = mBase + r + 8 * kh;
            const size_t idx = (size_t)m * N + n;
            const float cur = acc[t][r] + bv;
            if (FUSE_LIF) {
                float mo = mem[idx];
                float reset = (mo > TH_F) ? 1.0f : 0.0f;
                float mn = BETA_F * mo + cur - reset * TH_F;
                mem[idx] = mn;
                spk[idx] = (_Float16)((mn > TH_F) ? 1.0f : 0.0f);
            } else {
                C[idx] = cur;
            }
        }
    }
}

// ---------------- LIF layer 1 (elementwise; records spikes) ---------------
__global__ __launch_bounds__(256) void snn_lif1(
    const float* __restrict__ cur1, float* __restrict__ mem1,
    float* __restrict__ spike_rec, _Float16* __restrict__ spk1h, int n)
{
    int i = blockIdx.x * 256 + threadIdx.x;
    if (i < n) {
        float mo = mem1[i];
        float reset = (mo > TH_F) ? 1.0f : 0.0f;
        float mn = BETA_F * mo + cur1[i] - reset * TH_F;
        mem1[i] = mn;
        float s = (mn > TH_F) ? 1.0f : 0.0f;
        spike_rec[i] = s;            // f32 spike_record output
        spk1h[i] = (_Float16)s;      // f16 operand for GEMM2
    }
}

// ------------- layer 3: per-row dot(spk2, W3) + LIF + mean accum ----------
__global__ __launch_bounds__(256) void snn_lif3(
    const _Float16* __restrict__ spk2h, const float* __restrict__ W3,
    const float* __restrict__ b3, float* __restrict__ mem3,
    float* __restrict__ acc, float* __restrict__ out, int t, int steps)
{
    const int lane = threadIdx.x & 31;
    const int wave = threadIdx.x >> 5;
    const int row  = blockIdx.x * 8 + wave;
    const _Float16* s = spk2h + (size_t)row * NH;
    float p = 0.f;
    for (int k = lane; k < NH; k += 32) p += (float)s[k] * W3[k];
    for (int off = 16; off > 0; off >>= 1) p += __shfl_xor(p, off, 32);
    if (lane == 0) {
        float cur = p + b3[0];
        float mo = mem3[row];
        float reset = (mo > TH_F) ? 1.0f : 0.0f;
        float mn = BETA_F * mo + cur - reset * TH_F;
        mem3[row] = mn;
        float a = acc[row] + mn;
        acc[row] = a;
        if (t == steps - 1) out[row] = a / (float)steps;
    }
}

// --------------------------------------------------------------------------
extern "C" void kernel_launch(void* const* d_in, const int* in_sizes, int n_in,
                              void* d_out, int out_size, void* d_ws, size_t ws_size,
                              hipStream_t stream)
{
    (void)in_sizes; (void)n_in; (void)ws_size;
    const float* x     = (const float*)d_in[0];
    const float* gamma = (const float*)d_in[1];
    const float* bnb   = (const float*)d_in[2];
    const float* W1    = (const float*)d_in[3];
    const float* b1    = (const float*)d_in[4];
    const float* W2    = (const float*)d_in[5];
    const float* b2    = (const float*)d_in[6];
    const float* W3    = (const float*)d_in[7];
    const float* b3    = (const float*)d_in[8];
    // d_in[9] is num_steps on device; recover it from out_size instead.
    int steps = (out_size - NB) / (NB * NH);
    if (steps < 1) steps = 1;

    float* out       = (float*)d_out;           // (B,) mean membrane
    float* spike_rec = out + NB;                // (steps, B, H) f32

    // ---- workspace carve-up (≈71 MB total) ----
    char*  ws  = (char*)d_ws;
    size_t off = 0;
    auto carve = [&](size_t bytes) -> void* {
        void* p = ws + off;
        off = (off + bytes + 255) & ~(size_t)255;
        return p;
    };
    float*    scale = (float*)   carve(ND * sizeof(float));
    float*    shift = (float*)   carve(ND * sizeof(float));
    _Float16* a16   = (_Float16*)carve((size_t)NB * ND * sizeof(_Float16));
    _Float16* w1h   = (_Float16*)carve((size_t)NH * ND * sizeof(_Float16));
    _Float16* w2h   = (_Float16*)carve((size_t)NH * NH * sizeof(_Float16));
    float*    cur1  = (float*)   carve((size_t)NB * NH * sizeof(float));
    float*    mem1  = (float*)   carve((size_t)NB * NH * sizeof(float));
    float*    mem2  = (float*)   carve((size_t)NB * NH * sizeof(float));
    float*    mem3  = (float*)   carve(NB * sizeof(float));
    float*    accb  = (float*)   carve(NB * sizeof(float));
    _Float16* spk1h = (_Float16*)carve((size_t)NB * NH * sizeof(_Float16));
    _Float16* spk2h = (_Float16*)carve((size_t)NB * NH * sizeof(_Float16));

    // ---- zero the state ----
    (void)hipMemsetAsync(mem1, 0, (size_t)NB * NH * sizeof(float), stream);
    (void)hipMemsetAsync(mem2, 0, (size_t)NB * NH * sizeof(float), stream);
    (void)hipMemsetAsync(mem3, 0, NB * sizeof(float), stream);
    (void)hipMemsetAsync(accb, 0, NB * sizeof(float), stream);

    // ---- BN stats + operand conversion ----
    snn_bn_stats<<<ND, 256, 0, stream>>>(x, gamma, bnb, scale, shift);
    snn_norm_to_f16<<<(NB * ND + 255) / 256, 256, 0, stream>>>(x, scale, shift, a16);
    snn_cvt_f16<<<(NH * ND + 255) / 256, 256, 0, stream>>>(W1, w1h, NH * ND);
    snn_cvt_f16<<<(NH * NH + 255) / 256, 256, 0, stream>>>(W2, w2h, NH * NH);

    // ---- GEMM1: cur1 = x_norm @ W1^T + b1 (computed once) ----
    {
        dim3 grid(NH / 128, NB / 128);
        snn_wmma_gemm<false><<<grid, 256, 0, stream>>>(
            a16, w1h, b1, cur1, nullptr, nullptr, NB, NH, ND);
    }

    // ---- time loop ----
    for (int t = 0; t < steps; ++t) {
        snn_lif1<<<(NB * NH + 255) / 256, 256, 0, stream>>>(
            cur1, mem1, spike_rec + (size_t)t * NB * NH, spk1h, NB * NH);
        dim3 grid(NH / 128, NB / 128);
        snn_wmma_gemm<true><<<grid, 256, 0, stream>>>(
            spk1h, w2h, b2, nullptr, mem2, spk2h, NB, NH, NH);
        snn_lif3<<<NB / 8, 256, 0, stream>>>(
            spk2h, W3, b3, mem3, accb, out, t, steps);
    }
    (void)W3;
}
